// ResShift_ws_dist_171798692038
// MI455X (gfx1250) — matrix operand order repaired and checked by reading, not compiled, not run on hardware
//
#include <hip/hip_runtime.h>

// ---------------------------------------------------------------------------
// CDNA5 (gfx1250, wave32) implementation of the shift-quantized bottleneck.
// All channel contractions run on v_wmma_f32_16x16x32_bf16 (f32 accumulate).
// ---------------------------------------------------------------------------

typedef __bf16 bf16;
typedef __attribute__((ext_vector_type(8)))  __bf16 v8bf;
typedef __attribute__((ext_vector_type(16))) __bf16 v16bf;
typedef __attribute__((ext_vector_type(8)))  float  v8f;

#define EPSV 1e-5f

static __device__ inline v8f v8f_zero() {
    v8f z;
#pragma unroll
    for (int i = 0; i < 8; ++i) z[i] = 0.0f;
    return z;
}

static __device__ inline v16bf v16bf_zero() {
    v16bf z;
#pragma unroll
    for (int i = 0; i < 16; ++i) z[i] = (bf16)0.0f;
    return z;
}

// A-fragment (16x32 bf16, M = lane%16): VGPR pairs cover K = {half*8..+7} and
// {16+half*8..+7}  -> two 16-byte loads per lane.
static __device__ inline v16bf load_frag_a(const bf16* __restrict__ row, int kb, int half) {
    v8bf lo = *(const v8bf*)(row + kb + half * 8);
    v8bf hi = *(const v8bf*)(row + kb + half * 8 + 16);
    return __builtin_shufflevector(lo, hi, 0, 1, 2, 3, 4, 5, 6, 7, 8, 9, 10, 11, 12, 13, 14, 15);
}

// B-fragment (32x16 bf16, N = lane%16): K = {half*8..+7} and {16+half*8..+7}
// (same order as A) -> two 16-byte loads per lane.
static __device__ inline v16bf load_frag_b(const bf16* __restrict__ colp, int kb, int half) {
    v8bf lo = *(const v8bf*)(colp + kb + half * 8);
    v8bf hi = *(const v8bf*)(colp + kb + half * 8 + 16);
    return __builtin_shufflevector(lo, hi, 0, 1, 2, 3, 4, 5, 6, 7, 8, 9, 10, 11, 12, 13, 14, 15);
}

static __device__ inline v8f wmma_bf16(v16bf a, v16bf b, v8f c) {
    // (neg_a, A, neg_b, B, c_mod, C, reuse_a, reuse_b)
    return __builtin_amdgcn_wmma_f32_16x16x32_bf16(false, a, false, b, (short)0, c, false, false);
}

// ---------------------------------------------------------------------------
// Weight preprocessing: min/max + bucketized affine -> eff = sign * 2^shift
// ---------------------------------------------------------------------------

__global__ void minmax_kernel(const float* __restrict__ w, int n, float* __restrict__ mm) {
    __shared__ float smn[256], smx[256];
    int t = threadIdx.x;
    float mn = 3.4e38f, mx = -3.4e38f;
    for (int i = t; i < n; i += 256) {
        float v = w[i];
        mn = fminf(mn, v);
        mx = fmaxf(mx, v);
    }
    smn[t] = mn; smx[t] = mx;
    __syncthreads();
    for (int s = 128; s > 0; s >>= 1) {
        if (t < s) {
            smn[t] = fminf(smn[t], smn[t + s]);
            smx[t] = fmaxf(smx[t], smx[t + s]);
        }
        __syncthreads();
    }
    if (t == 0) { mm[0] = smn[0]; mm[1] = smx[0]; }
}

static __device__ inline float eff_val(float f, float mn, float mx,
                                       const float* __restrict__ shf,
                                       const float* __restrict__ sgf) {
    float t = (f - mn) * (100.0f / (mx - mn));
    int idx = (int)floorf(t);
    bool valid = (idx >= 0) && (idx < 100);
    int ic = min(max(idx, 0), 99);
    float shift = valid ? f * shf[ic] : 0.0f;
    float sign  = valid ? f * sgf[ic] : 0.0f;
    return sign * exp2f(shift);
}

__global__ void eff1x1_kernel(const float* __restrict__ w, const float* __restrict__ mm,
                              const float* __restrict__ shf, const float* __restrict__ sgf,
                              bf16* __restrict__ out, int n) {
    int idx = blockIdx.x * blockDim.x + threadIdx.x;
    if (idx >= n) return;
    out[idx] = (bf16)eff_val(w[idx], mm[0], mm[1], shf, sgf);
}

// w_c2: OIHW (128,128,3,3) -> out[tap][O=128][I=128] bf16
__global__ void effc2_kernel(const float* __restrict__ w, const float* __restrict__ mm,
                             const float* __restrict__ shf, const float* __restrict__ sgf,
                             bf16* __restrict__ out) {
    int idx = blockIdx.x * blockDim.x + threadIdx.x;
    if (idx >= 128 * 128 * 9) return;
    int tap = idx % 9;
    int oi  = idx / 9;           // = o*128 + i
    int i   = oi & 127;
    int o   = oi >> 7;
    out[(size_t)(tap * 128 + o) * 128 + i] = (bf16)eff_val(w[idx], mm[0], mm[1], shf, sgf);
}

// ---------------------------------------------------------------------------
// x: NCHW f32 -> NHWC bf16
// ---------------------------------------------------------------------------

__global__ void x_to_nhwc_kernel(const float* __restrict__ x, bf16* __restrict__ xb, int total) {
    int idx = blockIdx.x * blockDim.x + threadIdx.x;
    if (idx >= total) return;
    int col = idx >> 8;          // C = 256
    int c   = idx & 255;
    int n   = col / 3136;
    int s   = col - n * 3136;
    xb[idx] = (bf16)x[((size_t)n * 256 + c) * 3136 + s];
}

// ---------------------------------------------------------------------------
// 1x1 conv as GEMM:  Y[cols][M] (f32, NHWC)  =  W[M][K] x X[cols][K]
// One wave computes a 32(M) x 64(N) tile with 2x4 wmma accumulators.
// ---------------------------------------------------------------------------

__global__ __launch_bounds__(256) void gemm1x1_kernel(const bf16* __restrict__ W,
                                                      const bf16* __restrict__ X,
                                                      float* __restrict__ Y,
                                                      int M, int K, int cols) {
    int tid  = blockIdx.x * blockDim.x + threadIdx.x;
    int wid  = tid >> 5;
    int lane = threadIdx.x & 31;
    int half = lane >> 4;
    int r    = lane & 15;

    int mTiles = M >> 5;
    int nTiles = cols >> 6;
    if (wid >= mTiles * nTiles) return;     // wave-uniform

    int mT = wid % mTiles;
    int nT = wid / mTiles;
    int mBase   = mT << 5;
    int colBase = nT << 6;

    v8f acc[2][4];
#pragma unroll
    for (int mi = 0; mi < 2; ++mi)
#pragma unroll
        for (int ni = 0; ni < 4; ++ni) acc[mi][ni] = v8f_zero();

    const bf16* arow[2];
#pragma unroll
    for (int mi = 0; mi < 2; ++mi) arow[mi] = W + (size_t)(mBase + mi * 16 + r) * K;
    const bf16* bcol[4];
#pragma unroll
    for (int ni = 0; ni < 4; ++ni) bcol[ni] = X + (size_t)(colBase + ni * 16 + r) * K;

    for (int kb = 0; kb < K; kb += 32) {
        v16bf a[2], b[4];
#pragma unroll
        for (int mi = 0; mi < 2; ++mi) a[mi] = load_frag_a(arow[mi], kb, half);
#pragma unroll
        for (int ni = 0; ni < 4; ++ni) b[ni] = load_frag_b(bcol[ni], kb, half);
#pragma unroll
        for (int mi = 0; mi < 2; ++mi)
#pragma unroll
            for (int ni = 0; ni < 4; ++ni) acc[mi][ni] = wmma_bf16(a[mi], b[ni], acc[mi][ni]);
    }

    // C/D layout: lane%16 = N column, VGPR j -> M = half*8 + j  (contiguous)
#pragma unroll
    for (int mi = 0; mi < 2; ++mi)
#pragma unroll
        for (int ni = 0; ni < 4; ++ni) {
            int col = colBase + ni * 16 + r;
            float* p = Y + (size_t)col * M + mBase + mi * 16 + half * 8;
            *(float4*)p       = make_float4(acc[mi][ni][0], acc[mi][ni][1], acc[mi][ni][2], acc[mi][ni][3]);
            *(float4*)(p + 4) = make_float4(acc[mi][ni][4], acc[mi][ni][5], acc[mi][ni][6], acc[mi][ni][7]);
        }
}

// ---------------------------------------------------------------------------
// 3x3 conv (128->128, pad 1) as 9-tap implicit GEMM over NHWC bf16 input.
// ---------------------------------------------------------------------------

__global__ __launch_bounds__(256) void gemm3x3_kernel(const bf16* __restrict__ W9, // [9][128][128]
                                                      const bf16* __restrict__ X,  // [cols][128]
                                                      float* __restrict__ Y,       // [cols][128]
                                                      int cols) {
    const int M = 128, K = 128;
    int tid  = blockIdx.x * blockDim.x + threadIdx.x;
    int wid  = tid >> 5;
    int lane = threadIdx.x & 31;
    int half = lane >> 4;
    int r    = lane & 15;

    int mTiles = M >> 5;                  // 4
    int nTiles = cols >> 6;
    if (wid >= mTiles * nTiles) return;

    int mT = wid % mTiles;
    int nT = wid / mTiles;
    int mBase   = mT << 5;
    int colBase = nT << 6;

    v8f acc[2][4];
#pragma unroll
    for (int mi = 0; mi < 2; ++mi)
#pragma unroll
        for (int ni = 0; ni < 4; ++ni) acc[mi][ni] = v8f_zero();

    int coln[4], yy[4], xx[4];
#pragma unroll
    for (int ni = 0; ni < 4; ++ni) {
        int col = colBase + ni * 16 + r;
        int n   = col / 3136;
        int rem = col - n * 3136;
        int y   = rem / 56;
        coln[ni] = col;
        yy[ni]   = y;
        xx[ni]   = rem - y * 56;
    }

#pragma unroll
    for (int tap = 0; tap < 9; ++tap) {
        int dy = tap / 3 - 1;
        int dx = tap % 3 - 1;
        const bf16* Wt = W9 + (size_t)tap * 128 * 128;
        const bf16* arow[2];
#pragma unroll
        for (int mi = 0; mi < 2; ++mi) arow[mi] = Wt + (size_t)(mBase + mi * 16 + r) * K;

        bool valid[4];
        const bf16* bcol[4];
#pragma unroll
        for (int ni = 0; ni < 4; ++ni) {
            int y2 = yy[ni] + dy;
            int x2 = xx[ni] + dx;
            valid[ni] = ((unsigned)y2 < 56u) && ((unsigned)x2 < 56u);
            bcol[ni]  = X + (size_t)(coln[ni] + dy * 56 + dx) * K;
        }

        for (int kb = 0; kb < K; kb += 32) {
            v16bf a[2], b[4];
#pragma unroll
            for (int mi = 0; mi < 2; ++mi) a[mi] = load_frag_a(arow[mi], kb, half);
#pragma unroll
            for (int ni = 0; ni < 4; ++ni) {
                if (valid[ni]) b[ni] = load_frag_b(bcol[ni], kb, half);
                else           b[ni] = v16bf_zero();
            }
#pragma unroll
            for (int mi = 0; mi < 2; ++mi)
#pragma unroll
                for (int ni = 0; ni < 4; ++ni) acc[mi][ni] = wmma_bf16(a[mi], b[ni], acc[mi][ni]);
        }
    }

#pragma unroll
    for (int mi = 0; mi < 2; ++mi)
#pragma unroll
        for (int ni = 0; ni < 4; ++ni) {
            int col = coln[ni];
            float* p = Y + (size_t)col * M + mBase + mi * 16 + half * 8;
            *(float4*)p       = make_float4(acc[mi][ni][0], acc[mi][ni][1], acc[mi][ni][2], acc[mi][ni][3]);
            *(float4*)(p + 4) = make_float4(acc[mi][ni][4], acc[mi][ni][5], acc[mi][ni][6], acc[mi][ni][7]);
        }
}

// ---------------------------------------------------------------------------
// Depthwise 3x3 (PEG), NHWC f32 in -> NHWC f32 out, raw fp32 weights.
// ---------------------------------------------------------------------------

__global__ void peg_dw_kernel(const float* __restrict__ in, const float* __restrict__ wpeg,
                              float* __restrict__ out, int cols) {
    int idx = blockIdx.x * blockDim.x + threadIdx.x;
    if (idx >= cols * 128) return;
    int col = idx >> 7;
    int c   = idx & 127;
    int n   = col / 3136;
    int rem = col - n * 3136;
    int y   = rem / 56;
    int x   = rem - y * 56;
    float s = 0.0f;
#pragma unroll
    for (int ky = 0; ky < 3; ++ky) {
        int y2 = y + ky - 1;
        if ((unsigned)y2 >= 56u) continue;
#pragma unroll
        for (int kx = 0; kx < 3; ++kx) {
            int x2 = x + kx - 1;
            if ((unsigned)x2 >= 56u) continue;
            s += in[(size_t)(col + (ky - 1) * 56 + (kx - 1)) * 128 + c] * wpeg[c * 9 + ky * 3 + kx];
        }
    }
    out[idx] = s;
}

// ---------------------------------------------------------------------------
// BatchNorm: deterministic two-stage per-channel sum / sumsq reduction.
// ---------------------------------------------------------------------------

__global__ void stats_partial_kernel(const float* __restrict__ Y, float* __restrict__ part,
                                     int C, int cols) {
    int p = blockIdx.x;
    int t = threadIdx.x;
    int P = gridDim.x;
    int chunk = (cols + P - 1) / P;
    int c0 = p * chunk;
    int c1 = min(cols, c0 + chunk);
    float s1[2] = {0.0f, 0.0f}, s2[2] = {0.0f, 0.0f};
    for (int col = c0; col < c1; ++col) {
        const float* row = Y + (size_t)col * C;
#pragma unroll
        for (int u = 0; u < 2; ++u) {
            int c = t + u * 256;
            if (c < C) {
                float v = row[c];
                s1[u] += v;
                s2[u] += v * v;
            }
        }
    }
#pragma unroll
    for (int u = 0; u < 2; ++u) {
        int c = t + u * 256;
        if (c < C) {
            part[(size_t)p * 2 * C + c]     = s1[u];
            part[(size_t)p * 2 * C + C + c] = s2[u];
        }
    }
}

__global__ void stats_finalize_kernel(const float* __restrict__ part,
                                      const float* __restrict__ g, const float* __restrict__ b,
                                      float* __restrict__ scb, int C, int P, float invN) {
    int c = blockIdx.x * blockDim.x + threadIdx.x;
    if (c >= C) return;
    float s1 = 0.0f, s2 = 0.0f;
    for (int p = 0; p < P; ++p) {
        s1 += part[(size_t)p * 2 * C + c];
        s2 += part[(size_t)p * 2 * C + C + c];
    }
    float mean = s1 * invN;
    float var  = s2 * invN - mean * mean;
    float rs   = rsqrtf(var + EPSV);
    float A    = g[c] * rs;
    scb[c]     = A;
    scb[C + c] = b[c] - mean * A;
}

// normalize + ReLU -> bf16 NHWC (next conv input). C must be a power of two.
__global__ void norm_relu_bf16_kernel(const float* __restrict__ in, const float* __restrict__ scb,
                                      bf16* __restrict__ out, int C, int total) {
    int idx = blockIdx.x * blockDim.x + threadIdx.x;
    if (idx >= total) return;
    int c = idx & (C - 1);
    float v = fmaf(in[idx], scb[c], scb[C + c]);
    out[idx] = (bf16)fmaxf(v, 0.0f);
}

// normalize + ReLU -> f32 NHWC (residual branch)
__global__ void norm_relu_f32_kernel(const float* __restrict__ in, const float* __restrict__ scb,
                                     float* __restrict__ out, int C, int total) {
    int idx = blockIdx.x * blockDim.x + threadIdx.x;
    if (idx >= total) return;
    int c = idx & (C - 1);
    float v = fmaf(in[idx], scb[c], scb[C + c]);
    out[idx] = fmaxf(v, 0.0f);
}

// bn3 + residual add + ReLU, NHWC f32 -> NCHW f32 output
__global__ void final_kernel(const float* __restrict__ h5, const float* __restrict__ scb,
                             const float* __restrict__ res, float* __restrict__ out, int total) {
    int idx = blockIdx.x * blockDim.x + threadIdx.x;
    if (idx >= total) return;
    int col = idx >> 9;          // C = 512
    int c   = idx & 511;
    float v = fmaf(h5[idx], scb[c], scb[512 + c]) + res[idx];
    v = fmaxf(v, 0.0f);
    int n = col / 3136;
    int s = col - n * 3136;
    out[((size_t)n * 512 + c) * 3136 + s] = v;
}

// ---------------------------------------------------------------------------
// Host orchestration
// ---------------------------------------------------------------------------

extern "C" void kernel_launch(void* const* d_in, const int* in_sizes, int n_in,
                              void* d_out, int out_size, void* d_ws, size_t ws_size,
                              hipStream_t stream) {
    (void)in_sizes; (void)n_in; (void)out_size; (void)ws_size;

    const float* x      = (const float*)d_in[0];
    const float* w_ds   = (const float*)d_in[1];
    const float* w_c1   = (const float*)d_in[2];
    const float* w_peg  = (const float*)d_in[3];
    const float* w_c2   = (const float*)d_in[4];
    const float* w_c3   = (const float*)d_in[5];
    const float* a_ds_s = (const float*)d_in[6];
    const float* a_ds_g = (const float*)d_in[7];
    const float* a_c1_s = (const float*)d_in[8];
    const float* a_c1_g = (const float*)d_in[9];
    const float* a_c2_s = (const float*)d_in[10];
    const float* a_c2_g = (const float*)d_in[11];
    const float* a_c3_s = (const float*)d_in[12];
    const float* a_c3_g = (const float*)d_in[13];
    const float* g_ds   = (const float*)d_in[14];
    const float* b_ds   = (const float*)d_in[15];
    const float* g_1    = (const float*)d_in[16];
    const float* b_1    = (const float*)d_in[17];
    const float* g_2    = (const float*)d_in[18];
    const float* b_2    = (const float*)d_in[19];
    const float* g_3    = (const float*)d_in[20];
    const float* b_3    = (const float*)d_in[21];
    float* out = (float*)d_out;

    const int cols = 32 * 56 * 56;           // 100352
    const float invN = 1.0f / (float)cols;

    // workspace carve (256B aligned)
    char* ws = (char*)d_ws;
    size_t off = 0;
    auto take = [&](size_t bytes) -> char* {
        char* p = ws + off;
        off += (bytes + 255) & ~(size_t)255;
        return p;
    };
    bf16*  XB   = (bf16*)take((size_t)cols * 256 * sizeof(bf16));   // x NHWC bf16
    float* BIG  = (float*)take((size_t)cols * 512 * sizeof(float)); // y_ds / h_peg / h5
    float* RES  = (float*)take((size_t)cols * 512 * sizeof(float)); // residual
    float* F1   = (float*)take((size_t)cols * 128 * sizeof(float)); // h1 / h3
    bf16*  H2   = (bf16*)take((size_t)cols * 128 * sizeof(bf16));
    bf16*  H4   = (bf16*)take((size_t)cols * 128 * sizeof(bf16));
    bf16*  WDS  = (bf16*)take((size_t)512 * 256 * sizeof(bf16));
    bf16*  WC1  = (bf16*)take((size_t)128 * 256 * sizeof(bf16));
    bf16*  WC2  = (bf16*)take((size_t)9 * 128 * 128 * sizeof(bf16));
    bf16*  WC3  = (bf16*)take((size_t)512 * 128 * sizeof(bf16));
    float* MM   = (float*)take(8 * sizeof(float));
    float* PART = (float*)take((size_t)256 * 2 * 512 * sizeof(float));
    float* SCB  = (float*)take(2 * 512 * sizeof(float));

    auto blocks = [](long long n) { return (unsigned)((n + 255) / 256); };

    // ---- effective weights ----
    minmax_kernel<<<1, 256, 0, stream>>>(w_ds, 512 * 256, MM + 0);
    minmax_kernel<<<1, 256, 0, stream>>>(w_c1, 128 * 256, MM + 2);
    minmax_kernel<<<1, 256, 0, stream>>>(w_c2, 128 * 128 * 9, MM + 4);
    minmax_kernel<<<1, 256, 0, stream>>>(w_c3, 512 * 128, MM + 6);
    eff1x1_kernel<<<blocks(512 * 256), 256, 0, stream>>>(w_ds, MM + 0, a_ds_s, a_ds_g, WDS, 512 * 256);
    eff1x1_kernel<<<blocks(128 * 256), 256, 0, stream>>>(w_c1, MM + 2, a_c1_s, a_c1_g, WC1, 128 * 256);
    effc2_kernel<<<blocks(128 * 128 * 9), 256, 0, stream>>>(w_c2, MM + 4, a_c2_s, a_c2_g, WC2);
    eff1x1_kernel<<<blocks(512 * 128), 256, 0, stream>>>(w_c3, MM + 6, a_c3_s, a_c3_g, WC3, 512 * 128);

    // ---- x -> NHWC bf16 ----
    x_to_nhwc_kernel<<<blocks((long long)cols * 256), 256, 0, stream>>>(x, XB, cols * 256);

    // ---- downsample branch: conv1x1 512x256 -> BN -> ReLU -> RES ----
    gemm1x1_kernel<<<(16 * (cols >> 6) + 7) / 8, 256, 0, stream>>>(WDS, XB, BIG, 512, 256, cols);
    stats_partial_kernel<<<256, 256, 0, stream>>>(BIG, PART, 512, cols);
    stats_finalize_kernel<<<2, 256, 0, stream>>>(PART, g_ds, b_ds, SCB, 512, 256, invN);
    norm_relu_f32_kernel<<<blocks((long long)cols * 512), 256, 0, stream>>>(BIG, SCB, RES, 512, cols * 512);

    // ---- main branch ----
    // conv1 1x1 128x256 -> F1 (f32)
    gemm1x1_kernel<<<(4 * (cols >> 6) + 7) / 8, 256, 0, stream>>>(WC1, XB, F1, 128, 256, cols);
    // PEG depthwise 3x3 -> BIG (f32)
    peg_dw_kernel<<<blocks((long long)cols * 128), 256, 0, stream>>>(F1, w_peg, BIG, cols);
    // bn1 + relu -> H2 bf16
    stats_partial_kernel<<<256, 256, 0, stream>>>(BIG, PART, 128, cols);
    stats_finalize_kernel<<<1, 256, 0, stream>>>(PART, g_1, b_1, SCB, 128, 256, invN);
    norm_relu_bf16_kernel<<<blocks((long long)cols * 128), 256, 0, stream>>>(BIG, SCB, H2, 128, cols * 128);
    // conv2 3x3 implicit GEMM -> F1 (f32)
    gemm3x3_kernel<<<(4 * (cols >> 6) + 7) / 8, 256, 0, stream>>>(WC2, H2, F1, cols);
    // bn2 + relu -> H4 bf16
    stats_partial_kernel<<<256, 256, 0, stream>>>(F1, PART, 128, cols);
    stats_finalize_kernel<<<1, 256, 0, stream>>>(PART, g_2, b_2, SCB, 128, 256, invN);
    norm_relu_bf16_kernel<<<blocks((long long)cols * 128), 256, 0, stream>>>(F1, SCB, H4, 128, cols * 128);
    // conv3 1x1 512x128 -> BIG (f32)
    gemm1x1_kernel<<<(16 * (cols >> 6) + 7) / 8, 256, 0, stream>>>(WC3, H4, BIG, 512, 128, cols);
    // bn3 + residual + relu -> out (NCHW f32)
    stats_partial_kernel<<<256, 256, 0, stream>>>(BIG, PART, 512, cols);
    stats_finalize_kernel<<<2, 256, 0, stream>>>(PART, g_3, b_3, SCB, 512, 256, invN);
    final_kernel<<<blocks((long long)cols * 512), 256, 0, stream>>>(BIG, SCB, RES, out, cols * 512);
}
